// FFSNN_v2_12120397709818
// MI455X (gfx1250) — matrix operand order repaired and checked by reading, not compiled, not run on hardware
//
#include <hip/hip_runtime.h>
#include <hip/hip_bf16.h>
#include <cstdint>

// ---------------------------------------------------------------------------
// FFSNN forward on MI455X (gfx1250, wave32, WMMA + TDM).
//
//  * spikes are {0,1} -> exact in bf16; weights split hi/lo bf16 so every
//    GEMM runs as 2x v_wmma_f32_16x16x32_bf16 with f32 accumulation
//    (~fp32 accuracy at bf16 matrix-core speed). hi and lo products use
//    INDEPENDENT accumulator chains (summed at the end) so back-to-back
//    WMMAs never RAW-depend on the same D registers.
//  * A tiles staged to LDS by the Tensor Data Mover (tensor_load_to_lds),
//    double-buffered, one issuing wave, s_wait_tensorcnt + barrier.
//    Descriptor padding (16 dwords data + 4 dwords pad) gives an 80B row
//    pitch -> conflict-free 16B fragment reads. Manual ds-staging fallback
//    if the builtin is unavailable.
//  * masks cycle>=2 => spike_prev==0 at every active update => mem=mem*0.5+drive.
//  * output layer is linear in s3 => single GEMM vs accumulated ss3 at end.
//  * per-timestep kernels (global all-to-all dependence); graph replay
//    amortizes launches. W hi/lo fragment buffers are L2-resident (8MB<<192MB).
// ---------------------------------------------------------------------------

typedef __attribute__((ext_vector_type(16))) __bf16 v16bf;
typedef __attribute__((ext_vector_type(8)))  float  v8f;

#define THRESH 0.5f
#define DECAY  0.5f
#define TSTEPS 98
#define HDIM   1024
#define NBATCH 1024

// ---- TDM availability (device pass only; host pass uses fallback body) ----
#if defined(__gfx1250__) && defined(__has_builtin)
# if __has_builtin(__builtin_amdgcn_tensor_load_to_lds) && \
     __has_builtin(__builtin_amdgcn_s_wait_tensorcnt)
#  define USE_TDM 1
# endif
#endif
#ifndef USE_TDM
# define USE_TDM 0
#endif

#if USE_TDM
typedef unsigned int v4u __attribute__((ext_vector_type(4)));
typedef int          v4i __attribute__((ext_vector_type(4)));
typedef int          v8i __attribute__((ext_vector_type(8)));

// Issue a 2-D TDM load: 128 rows x 32 bf16 tile out of a [.,1024] bf16
// matrix into LDS, with 16B padding appended after every 64B of row data.
__device__ __forceinline__ void tdm_load_tile(uint32_t lds_byte,
                                              const __bf16* gptr) {
  uint64_t ga = (uint64_t)(uintptr_t)gptr;
  v4u g0;
  g0.x = 1u;                                   // count=1 (valid user D#)
  g0.y = lds_byte;                             // lds_addr
  g0.z = (uint32_t)ga;                         // global_addr[31:0]
  g0.w = (uint32_t)((ga >> 32) & 0x1FFFFFFu)   // global_addr[56:32]
         | (2u << 30);                         // type = 2 ("image")
  v8i g1;
  g1[0] = (1 << 16)                            // data_size = 2 bytes
        | (1 << 20)                            // pad_enable
        | (3 << 22)                            // pad_interval = 16 dwords
        | (3 << 25);                           // pad_amount   = 4 dwords
  g1[1] = (int)(1024u << 16);                  // tensor_dim0[15:0] = 1024
  g1[2] = (int)(1024u << 16);                  // tensor_dim1[15:0] = 1024
  g1[3] = (int)(32u << 16);                    // tile_dim0 = 32
  g1[4] = 128;                                 // tile_dim1 = 128, tile_dim2 = 0
  g1[5] = 1024;                                // tensor_dim0_stride = 1024
  g1[6] = 0;
  g1[7] = 0;
  v4i z4 = {0, 0, 0, 0};
#if __clang_major__ >= 23
  v8i z8 = {0, 0, 0, 0, 0, 0, 0, 0};
  __builtin_amdgcn_tensor_load_to_lds(g0, g1, z4, z4, z8, 0);
#else
  __builtin_amdgcn_tensor_load_to_lds(g0, g1, z4, z4, 0);
#endif
}
#endif  // USE_TDM

// ---------------------------------------------------------------------------
// Pre-swizzle a 1024x1024 f32 weight matrix (row-major [j][k]) into
// WMMA B-fragment-linear hi/lo bf16 buffers.
// B tile = 32(K) x 16(N); lane = 16*((k/8)%2) + (j%16), elem = 8*(k/16) + k%8.
// Flat: (((kt*64 + nt)*32 + lane)*16 + e) -> each lane's 16 bf16 contiguous.
// ---------------------------------------------------------------------------
__global__ __launch_bounds__(256) void build_bfrag(const float* __restrict__ W,
                                                   __bf16* __restrict__ hi,
                                                   __bf16* __restrict__ lo) {
  int gid = blockIdx.x * 256 + threadIdx.x;          // 0 .. 1048575
  int e    = gid & 15;
  int lane = (gid >> 4) & 31;
  int nt   = (gid >> 9) & 63;
  int kt   = gid >> 15;
  int k = kt * 32 + ((e >> 3) << 4) + ((lane >> 4) << 3) + (e & 7);
  int j = nt * 16 + (lane & 15);
  float w = W[(size_t)j * HDIM + k];
  __bf16 h = (__bf16)w;
  hi[gid] = h;
  lo[gid] = (__bf16)(w - (float)h);
}

// ---------------------------------------------------------------------------
// Layer 1 (K=8, tiny): one thread per (n,h).
// ---------------------------------------------------------------------------
__global__ __launch_bounds__(256) void snn_layer1(const float* __restrict__ x,
                                                  const float* __restrict__ W1,
                                                  const float* __restrict__ b1,
                                                  const int*   __restrict__ mask1,
                                                  float* __restrict__ mem,
                                                  float* __restrict__ ss,
                                                  __bf16* __restrict__ sdst,
                                                  int t, int start) {
  int gid = blockIdx.x * 256 + threadIdx.x;          // 0 .. 1048575
  int n = gid >> 10;
  int h = gid & (HDIM - 1);
  const float* xr = x + (size_t)n * 784 + start;
  const float* wr = W1 + h * 8;
  float d = b1[h];
#pragma unroll
  for (int k = 0; k < 8; ++k) d += xr[k] * wr[k];
  size_t idx = ((size_t)n << 10) + h;
  float sp = 0.0f;
  if (mask1[h * TSTEPS + t]) {
    float nm = mem[idx] * DECAY + d;                 // spike_prev provably 0
    mem[idx] = nm;
    sp = (nm > THRESH) ? 1.0f : 0.0f;
    if (sp != 0.0f) ss[idx] += 1.0f;
  }
  sdst[idx] = (__bf16)sp;
}

// ---------------------------------------------------------------------------
// Fused hidden-layer step: D = spikes_in @ W^T (1024^3, bf16 hi/lo WMMA)
// -> + bias -> masked membrane update -> spike -> ss accumulate -> bf16
// spike store for next layer (snext null for layer 3).
// Block = 256 thr (8 waves), tile 128x128; wave tile 64x32 (4x2 accums x2).
// ---------------------------------------------------------------------------
__global__ __launch_bounds__(256) void snn_gemm_layer(const __bf16* __restrict__ Ain,
                                                      const __bf16* __restrict__ Bhi,
                                                      const __bf16* __restrict__ Blo,
                                                      const float*  __restrict__ bias,
                                                      const int*    __restrict__ mask,
                                                      float* __restrict__ mem,
                                                      float* __restrict__ ss,
                                                      __bf16* __restrict__ snext,
                                                      int t) {
  const int tid   = threadIdx.x;
  const int lane  = tid & 31;
  const int wid   = tid >> 5;
  const int waveM = wid & 1;        // 2 wave rows  (64 each)
  const int waveN = wid >> 1;       // 4 wave cols  (32 each)
  const int bm = blockIdx.x;        // M tile (batch)
  const int bn = blockIdx.y;        // N tile (hidden out)

  // Independent hi/lo accumulator chains -> no WMMA->WMMA RAW on D.
  v8f accH[4][2], accL[4][2];
  const v8f vzero = {0.f, 0.f, 0.f, 0.f, 0.f, 0.f, 0.f, 0.f};
#pragma unroll
  for (int i = 0; i < 4; ++i)
#pragma unroll
    for (int j = 0; j < 2; ++j) { accH[i][j] = vzero; accL[i][j] = vzero; }

  union Frag { uint4 u[2]; v16bf v; };
  const int h16  = lane >> 4;
  const int mrow = lane & 15;

#if USE_TDM
  // Double-buffered A tiles filled by the Tensor Data Mover.
  // Row pitch 80B (64B data + 16B pad from the descriptor) = 5 uint4.
  __shared__ uint4 ldsA[2][128 * 5];
  const __bf16* tileBase = Ain + (((size_t)bm) << 17);   // bm*128*1024
  uint32_t lds0 = (uint32_t)(uintptr_t)&ldsA[0][0];
  uint32_t lds1 = (uint32_t)(uintptr_t)&ldsA[1][0];
  if (wid == 0) tdm_load_tile(lds0, tileBase);
#else
  __shared__ uint4 ldsA[1][128 * 5];
  const int rA = tid >> 1;          // staging row (0..127)
  const int cA = tid & 1;           // staging 16-elem chunk
#endif

  for (int kt = 0; kt < 32; ++kt) {
#if USE_TDM
    if (wid == 0) __builtin_amdgcn_s_wait_tensorcnt(0);  // tile kt landed
    __syncthreads();                                     // publish + buffer safe
    if (wid == 0 && kt + 1 < 32)                         // overlap DMA kt+1
      tdm_load_tile((kt & 1) ? lds0 : lds1, tileBase + (kt + 1) * 32);
    const uint4* tile = &ldsA[kt & 1][0];
#else
    {
      const uint4* src = (const uint4*)(Ain + (((size_t)(bm * 128 + rA)) << 10) +
                                        kt * 32 + cA * 16);
      uint4 a0 = src[0];
      uint4 a1 = src[1];
      ldsA[0][rA * 5 + cA * 2]     = a0;
      ldsA[0][rA * 5 + cA * 2 + 1] = a1;
      if (kt + 1 < 32)
        __builtin_prefetch((const void*)(Ain + (((size_t)(bm * 128 + rA)) << 10) +
                                         (kt + 1) * 32), 0, 1);
    }
    __syncthreads();
    const uint4* tile = &ldsA[0][0];
#endif

    // ---- A fragments (per ISA 16-bit 16x32 layout) ----
    Frag af[4];
#pragma unroll
    for (int fm = 0; fm < 4; ++fm) {
      int row = waveM * 64 + fm * 16 + mrow;
      af[fm].u[0] = tile[row * 5 + h16];        // K = h16*8 .. +7
      af[fm].u[1] = tile[row * 5 + 2 + h16];    // K = 16 + h16*8 .. +7
    }

    // ---- B fragments (pre-swizzled, lane-contiguous) ----
    Frag bh[2], bl[2];
#pragma unroll
    for (int fn = 0; fn < 2; ++fn) {
      int nt = bn * 8 + waveN * 2 + fn;
      size_t boff = ((size_t)(kt * 64 + nt) << 9) + lane * 16;
      const uint4* ph = (const uint4*)(Bhi + boff);
      const uint4* pl = (const uint4*)(Blo + boff);
      bh[fn].u[0] = ph[0]; bh[fn].u[1] = ph[1];
      bl[fn].u[0] = pl[0]; bl[fn].u[1] = pl[1];
    }

    // ---- 16 WMMAs over 16 independent accumulator chains ----
#pragma unroll
    for (int fn = 0; fn < 2; ++fn)
#pragma unroll
      for (int fm = 0; fm < 4; ++fm)
        accH[fm][fn] = __builtin_amdgcn_wmma_f32_16x16x32_bf16(
            false, af[fm].v, false, bh[fn].v, (short)0, accH[fm][fn], false, false);
#pragma unroll
    for (int fn = 0; fn < 2; ++fn)
#pragma unroll
      for (int fm = 0; fm < 4; ++fm)
        accL[fm][fn] = __builtin_amdgcn_wmma_f32_16x16x32_bf16(
            false, af[fm].v, false, bl[fn].v, (short)0, accL[fm][fn], false, false);

#if !USE_TDM
    __syncthreads();
#endif
  }

  // ---- epilogue: bias + masked membrane update + spike + accumulators ----
  // D layout: VGPR r, lane -> (M = r + 8*(lane/16), N = lane%16)
  const int dh = lane >> 4;
  const int nc = lane & 15;
#pragma unroll
  for (int fn = 0; fn < 2; ++fn) {
    int j = bn * 128 + (waveN * 2 + fn) * 16 + nc;
    int mk = mask[j * TSTEPS + t];
    float bj = bias[j];
#pragma unroll
    for (int fm = 0; fm < 4; ++fm) {
      v8f acc = accH[fm][fn] + accL[fm][fn];
      int n0 = bm * 128 + waveM * 64 + fm * 16 + dh * 8;
#pragma unroll
      for (int r = 0; r < 8; ++r) {
        size_t idx = (((size_t)(n0 + r)) << 10) + j;
        float sp = 0.0f;
        if (mk) {
          float nm = mem[idx] * DECAY + acc[r] + bj;
          mem[idx] = nm;
          sp = (nm > THRESH) ? 1.0f : 0.0f;
          if (sp != 0.0f) ss[idx] += 1.0f;
        }
        if (snext) snext[idx] = (__bf16)sp;
      }
    }
  }
}

// ---------------------------------------------------------------------------
// Final kernels: scale ss by 1/T; deterministic two-stage firing-rate
// reduction; output GEMM vs accumulated s3 sums.
// ---------------------------------------------------------------------------
__global__ __launch_bounds__(256) void scale_ss(float* __restrict__ ss) {
  size_t gid = (size_t)blockIdx.x * 256 + threadIdx.x;   // 3*1048576 elements
  ss[gid] *= (1.0f / (float)TSTEPS);
}

__global__ __launch_bounds__(256) void fr_partial(const float* __restrict__ ssbase,
                                                  float* __restrict__ partial) {
  __shared__ float red[256];
  int layer = blockIdx.y;
  const float* p = ssbase + (size_t)layer * (1u << 20) +
                   (size_t)blockIdx.x * 16384 + (size_t)threadIdx.x * 64;
  float s = 0.0f;
#pragma unroll 4
  for (int i = 0; i < 64; ++i) s += p[i];
  red[threadIdx.x] = s;
  __syncthreads();
  for (int w = 128; w > 0; w >>= 1) {
    if (threadIdx.x < w) red[threadIdx.x] += red[threadIdx.x + w];
    __syncthreads();
  }
  if (threadIdx.x == 0) partial[layer * 64 + blockIdx.x] = red[0];
}

__global__ void fr_final(const float* __restrict__ partial, float* __restrict__ fr) {
  int l = threadIdx.x;
  if (l < 3) {
    float s = 0.0f;
    for (int b = 0; b < 64; ++b) s += partial[l * 64 + b];
    fr[l] = s * (1.0f / ((float)NBATCH * (float)HDIM));  // ss already /T
  }
}

__global__ __launch_bounds__(256) void out_gemm(const float* __restrict__ ss3,
                                                const float* __restrict__ W4,
                                                const float* __restrict__ b4,
                                                float* __restrict__ out) {
  int gid = blockIdx.x * 256 + threadIdx.x;   // 10240
  if (gid >= NBATCH * 10) return;
  int n = gid / 10, o = gid % 10;
  const float* sr = ss3 + ((size_t)n << 10);
  const float* wr = W4 + (size_t)o * HDIM;
  float s = 0.0f;
#pragma unroll 8
  for (int h = 0; h < HDIM; ++h) s += sr[h] * wr[h];
  out[gid] = s + b4[o];   // (Sum_t s3/T) @ W4^T + b4  ==  osum/T
}

// ---------------------------------------------------------------------------
extern "C" void kernel_launch(void* const* d_in, const int* in_sizes, int n_in,
                              void* d_out, int out_size, void* d_ws, size_t ws_size,
                              hipStream_t stream) {
  const float* input = (const float*)d_in[0];
  const float* W1    = (const float*)d_in[1];
  const float* b1    = (const float*)d_in[2];
  const float* W2    = (const float*)d_in[3];
  const float* b2    = (const float*)d_in[4];
  const float* W3    = (const float*)d_in[5];
  const float* b3    = (const float*)d_in[6];
  const float* W4    = (const float*)d_in[7];
  const float* b4    = (const float*)d_in[8];
  const int* mask1   = (const int*)d_in[9];
  const int* mask2   = (const int*)d_in[10];
  const int* mask3   = (const int*)d_in[11];
  float* out = (float*)d_out;

  const size_t MB20 = 1u << 20;              // 1M elements
  // workspace layout (~24 MB)
  float*  m1   = (float*)d_ws;               // 1M f32
  float*  m2   = m1 + MB20;
  float*  m3   = m2 + MB20;
  __bf16* s1d  = (__bf16*)(m3 + MB20);       // 1M bf16
  __bf16* s2d  = s1d + MB20;
  __bf16* B2hi = s2d + MB20;                 // 1M bf16 each
  __bf16* B2lo = B2hi + MB20;
  __bf16* B3hi = B2lo + MB20;
  __bf16* B3lo = B3hi + MB20;
  float*  part = (float*)(B3lo + MB20);      // 192 f32

  // d_out layout: outputs[10240] | ss1 | ss2 | ss3 (1M each) | fr[3]
  float* ss1 = out + 10240;
  float* ss2 = ss1 + MB20;
  float* ss3 = ss2 + MB20;
  float* fr  = out + ((size_t)out_size - 3);

  // zero accumulators / state (graph-capturable stream memsets)
  hipMemsetAsync(d_out, 0, (size_t)out_size * sizeof(float), stream);
  hipMemsetAsync(m1, 0, 3 * MB20 * sizeof(float), stream);

  // pre-swizzle weights into WMMA B-fragment hi/lo layout (L2-resident after)
  build_bfrag<<<4096, 256, 0, stream>>>(W2, B2hi, B2lo);
  build_bfrag<<<4096, 256, 0, stream>>>(W3, B3hi, B3lo);

  dim3 ggrid(8, 8);  // 1024/128 x 1024/128
  for (int t = 0; t < TSTEPS; ++t) {
    // faithful replication of the reference's (buggy) slice start
    int start = (t * 8 < TSTEPS - 8) ? t * 8 : (784 - 8);
    snn_layer1<<<4096, 256, 0, stream>>>(input, W1, b1, mask1, m1, ss1, s1d,
                                         t, start);
    snn_gemm_layer<<<ggrid, 256, 0, stream>>>(s1d, B2hi, B2lo, b2, mask2,
                                              m2, ss2, s2d, t);
    snn_gemm_layer<<<ggrid, 256, 0, stream>>>(s2d, B3hi, B3lo, b3, mask3,
                                              m3, ss3, (__bf16*)nullptr, t);
  }

  scale_ss<<<12288, 256, 0, stream>>>(ss1);          // 3M elements /= T
  fr_partial<<<dim3(64, 3), 256, 0, stream>>>(ss1, part);
  fr_final<<<1, 64, 0, stream>>>(part, fr);
  out_gemm<<<40, 256, 0, stream>>>(ss3, W4, b4, out);
}
